// FastAttention_5274219839587
// MI455X (gfx1250) — matrix operand order repaired and checked by reading, not compiled, not run on hardware
//
#include <hip/hip_runtime.h>

// ---------------------------------------------------------------------------
// Performer fast-attention for MI455X (gfx1250), wave32 + bf16 WMMA.
//   out = (ratio*(exp(qdash - qdiag - rowmax)+EPS)) @ (ratio*(exp(kdash - kdiag - gmax)+EPS))^T-contracted-with-v
// Factored so EPS / gmax / rowmax are applied exactly in f32 outside the GEMMs.
// ---------------------------------------------------------------------------

typedef __bf16 bf16;
typedef bf16  v16bf __attribute__((ext_vector_type(16)));
typedef bf16  v8bf  __attribute__((ext_vector_type(8)));
typedef float v8f   __attribute__((ext_vector_type(8)));
typedef float v4f   __attribute__((ext_vector_type(4)));

#define BH_     64          // B*H
#define N_      4096
#define D_      64
#define M_      256
#define NCH_    4           // k-chunks per (b,h)
#define CHROWS_ 1024        // N_/NCH_
#define NORM2_      0.125f                  // normalizer^2 = 1/sqrt(D)
#define NORMALIZER_ 0.35355339059327373f    // D^-0.25
#define RATIO_      0.0625f                 // M^-0.5
#define EPSV_       1e-4f

static __device__ __forceinline__ v8f wmma_bf16(v16bf a, v16bf b, v8f c) {
  // D = A(16x32 bf16) x B(32x16 bf16) + C(16x16 f32)
  return __builtin_amdgcn_wmma_f32_16x16x32_bf16(false, a, false, b, (short)0, c,
                                                 false, false);
}
static __device__ __forceinline__ v16bf cat16(v8bf lo, v8bf hi) {
  return __builtin_shufflevector(lo, hi, 0,1,2,3,4,5,6,7,8,9,10,11,12,13,14,15);
}
// A fragment (16-bit A 16x32): lane holds row (lane&15); its 16
// K-halves live at [hi*16 .. hi*16+15] contiguous (hi = lane>>4), matching B.
// rowptr points at the K=0 half of this lane's row. 16B-aligned chunks.
static __device__ __forceinline__ v16bf a_frag(const bf16* rowptr, int hi) {
  const v8bf* p0 = (const v8bf*)(rowptr + hi * 16);
  const v8bf* p1 = (const v8bf*)(rowptr + hi * 16 + 8);
  return cat16(*p0, *p1);
}
// B fragment (16-bit B 32x16): lane holds column (lane&15); 16 contiguous
// K-halves starting at K = hi*16. p points at that first half (16B aligned).
static __device__ __forceinline__ v16bf b_frag(const bf16* p) {
  return cat16(*(const v8bf*)p, *(const v8bf*)(p + 8));
}

// ---------------------------------------------------------------------------
// Kernel 0: proj (fp32 [M,D]) -> bf16, L2-resident operand for both GEMM1s.
// ---------------------------------------------------------------------------
__global__ void k_prep(const float* __restrict__ proj, bf16* __restrict__ projb) {
  int i = blockIdx.x * 256 + threadIdx.x;      // grid 64 * 256 = 16384 = M*D
  projb[i] = (bf16)proj[i];
}

// ---------------------------------------------------------------------------
// Kernel 1: per (bh, chunk): Craw[m,e] = sum_n exp(kdash - kdiag) * v[n,e],
// plus chunk max(kdash) and vsum[e]. One pass over k/v; bf16 WMMA everywhere.
// ---------------------------------------------------------------------------
__launch_bounds__(256)
__global__ void k_ctx(const float* __restrict__ kmat, const float* __restrict__ vmat,
                      const bf16* __restrict__ projb,
                      float* __restrict__ ctxpart, float* __restrict__ vsumpart,
                      float* __restrict__ chunkmax) {
  __shared__ __attribute__((aligned(32))) bf16  Klds[32 * 64];   // k tile, row-major
  __shared__ __attribute__((aligned(32))) bf16  VT[64 * 40];     // v^T, padded stride
  __shared__ __attribute__((aligned(32))) bf16  ET[256 * 40];    // E^T, padded stride
  __shared__ __attribute__((aligned(32))) float diag[32];
  __shared__ float redf[8];
  __shared__ float vsp[4][64];

  const int t = threadIdx.x, lane = t & 31, w = t >> 5;
  const int lo = lane & 15, hi = lane >> 4;
  const int bh = blockIdx.x, ch = blockIdx.y;
  const float* kb_ = kmat + (size_t)bh * N_ * D_ + (size_t)ch * CHROWS_ * D_;
  const float* vb_ = vmat + (size_t)bh * N_ * D_ + (size_t)ch * CHROWS_ * D_;

  const int ns = w & 1;        // n sub-tile (0/1) for dash phase
  const int mg = w >> 1;       // m group (4 m-tiles) for dash phase

  v8f cacc[2][4];
  #pragma unroll
  for (int a = 0; a < 2; ++a)
    #pragma unroll
    for (int b = 0; b < 4; ++b) cacc[a][b] = (v8f){0,0,0,0,0,0,0,0};
  float vsum_priv = 0.f;
  float mx = -3.4e38f;

  for (int s = 0; s < CHROWS_ / 32; ++s) {
    const float* kstep = kb_ + s * 32 * D_;
    const float* vstep = vb_ + s * 32 * D_;
    if (s + 1 < CHROWS_ / 32) {                    // CDNA5 global_prefetch
      __builtin_prefetch(kstep + 32 * D_ + t * 8, 0, 1);
      __builtin_prefetch(vstep + 32 * D_ + t * 8, 0, 1);
    }
    // ---- stage: k -> Klds (bf16), v -> VT (bf16 transposed), vsum ----
    #pragma unroll
    for (int i = 0; i < 8; ++i) {
      int li = t + 256 * i;                        // row = li>>6 = (t>>6)+4i, d = t&63
      float kv = kstep[li];
      Klds[li] = (bf16)kv;
      float vv = vstep[li];
      VT[(li & 63) * 40 + (li >> 6)] = (bf16)vv;
      vsum_priv += vv;                             // fixed e = t&63 per thread
    }
    // ---- diag[n] = 0.5*norm2 * sum_d k^2 (f32, from global) ----
    {
      int row = t >> 3, d0 = (t & 7) * 8;
      const float* p = kstep + row * 64 + d0;
      float s2 = 0.f;
      #pragma unroll
      for (int j = 0; j < 8; ++j) { float x = p[j]; s2 += x * x; }
      s2 += __shfl_xor(s2, 1); s2 += __shfl_xor(s2, 2); s2 += __shfl_xor(s2, 4);
      if ((t & 7) == 0) diag[row] = s2 * (0.5f * NORM2_);
    }
    __syncthreads();
    // ---- dash = normalizer * k @ proj^T ; E = exp(dash - diag) -> ET ----
    {
      const bf16* arow = Klds + (ns * 16 + lo) * 64;
      v16bf a0 = a_frag(arow, hi);                 // K = d 0..31
      v16bf a1 = a_frag(arow + 32, hi);            // K = d 32..63
      v4f dg0 = *(const v4f*)&diag[ns * 16 + hi * 8];
      v4f dg1 = *(const v4f*)&diag[ns * 16 + hi * 8 + 4];
      #pragma unroll
      for (int mt = 0; mt < 4; ++mt) {
        int mcol = (mg * 4 + mt) * 16 + lo;
        const bf16* bp = projb + mcol * 64 + hi * 16;
        v8f acc = (v8f){0,0,0,0,0,0,0,0};
        acc = wmma_bf16(a0, b_frag(bp), acc);
        acc = wmma_bf16(a1, b_frag(bp + 32), acc);
        #pragma unroll
        for (int r = 0; r < 8; ++r) {
          float dv = acc[r] * NORMALIZER_;
          mx = fmaxf(mx, dv);
          float dgv = (r < 4) ? dg0[r] : dg1[r - 4];
          float ev = __expf(dv - dgv);
          ET[mcol * 40 + ns * 16 + r + 8 * hi] = (bf16)ev;
        }
      }
    }
    __syncthreads();
    // ---- ctx accumulate: Craw[m,e] += E^T @ v  (wave owns m rows w*32..+31) ----
    #pragma unroll
    for (int mt2 = 0; mt2 < 2; ++mt2) {
      const bf16* arow = ET + (w * 32 + mt2 * 16 + lo) * 40;
      v16bf a = a_frag(arow, hi);                  // K = n 0..31
      #pragma unroll
      for (int et = 0; et < 4; ++et) {
        const bf16* bp = VT + (et * 16 + lo) * 40 + hi * 16;
        cacc[mt2][et] = wmma_bf16(a, b_frag(bp), cacc[mt2][et]);
      }
    }
    __syncthreads();
  }

  // ---- epilogue: partial context, chunk max, vsum partials ----
  float* cp = ctxpart + (size_t)(bh * NCH_ + ch) * M_ * D_;
  #pragma unroll
  for (int mt2 = 0; mt2 < 2; ++mt2)
    #pragma unroll
    for (int et = 0; et < 4; ++et)
      #pragma unroll
      for (int r = 0; r < 8; ++r) {
        int m = w * 32 + mt2 * 16 + r + 8 * hi;
        int e = et * 16 + lo;
        cp[m * 64 + e] = cacc[mt2][et][r];
      }
  mx = fmaxf(mx, __shfl_xor(mx, 16)); mx = fmaxf(mx, __shfl_xor(mx, 8));
  mx = fmaxf(mx, __shfl_xor(mx, 4));  mx = fmaxf(mx, __shfl_xor(mx, 2));
  mx = fmaxf(mx, __shfl_xor(mx, 1));
  if (lane == 0) redf[w] = mx;
  vsp[t >> 6][t & 63] = vsum_priv;
  __syncthreads();
  if (t == 0) {
    float g = redf[0];
    #pragma unroll
    for (int i = 1; i < 8; ++i) g = fmaxf(g, redf[i]);
    chunkmax[bh * NCH_ + ch] = g;
  }
  if (t < 64)
    vsumpart[(bh * NCH_ + ch) * 64 + t] = vsp[0][t] + vsp[1][t] + vsp[2][t] + vsp[3][t];
}

// ---------------------------------------------------------------------------
// Kernel 2: gmax over all chunks; ctx = ratio*(e^-gmax * sum Craw + EPS*vsum);
// emit ctx^T as bf16 [e][m] plus ctxsum[e] = sum_m ctx[m][e] (exact f32).
// ---------------------------------------------------------------------------
__global__ void k_red(const float* __restrict__ ctxpart, const float* __restrict__ vsumpart,
                      const float* __restrict__ chunkmax,
                      bf16* __restrict__ ctxT, float* __restrict__ ctxsum) {
  __shared__ float csp[4][64];
  const int bh = blockIdx.x, t = threadIdx.x;
  float g = chunkmax[0];
  for (int i = 1; i < BH_ * NCH_; ++i) g = fmaxf(g, chunkmax[i]);   // global max (all bh)
  const float esc = __expf(-g);
  const float* cp = ctxpart + (size_t)bh * NCH_ * M_ * D_;
  const float* vp = vsumpart + bh * NCH_ * 64;
  float psum = 0.f;
  for (int i = 0; i < 64; ++i) {
    int idx = t + 256 * i;                  // m = idx>>6, e = idx&63 (e fixed per thread)
    int m = idx >> 6, e = idx & 63;
    float craw = cp[idx] + cp[idx + M_ * D_] + cp[idx + 2 * M_ * D_] + cp[idx + 3 * M_ * D_];
    float vs = vp[e] + vp[64 + e] + vp[128 + e] + vp[192 + e];
    float val = RATIO_ * (esc * craw + EPSV_ * vs);
    ctxT[(size_t)bh * D_ * M_ + (size_t)e * M_ + m] = (bf16)val;
    psum += val;
  }
  csp[t >> 6][t & 63] = psum;
  __syncthreads();
  if (t < 64) ctxsum[bh * 64 + t] = csp[0][t] + csp[1][t] + csp[2][t] + csp[3][t];
}

// ---------------------------------------------------------------------------
// Kernel 3: per 16-row q tile: qdash GEMM (16 tiles live in VGPRs), row max via
// shfl within 16-lane groups, exp, then qp @ ctx^T GEMM. EPS applied exactly:
// out = ratio*(O + EPS*ctxsum[e]).
// ---------------------------------------------------------------------------
__launch_bounds__(256)
__global__ void k_out(const float* __restrict__ q, const bf16* __restrict__ projb,
                      const bf16* __restrict__ ctxT, const float* __restrict__ ctxsum,
                      float* __restrict__ out) {
  __shared__ __attribute__((aligned(32))) bf16  Q[8][16 * 64];
  __shared__ __attribute__((aligned(32))) bf16  QP[8][16 * 40];   // padded stride 40
  __shared__ __attribute__((aligned(32))) float DG[8][16];

  const int t = threadIdx.x, lane = t & 31, w = t >> 5;
  const int lo = lane & 15, hi = lane >> 4;
  const int bh = blockIdx.x;
  const int n0 = blockIdx.y * 128 + w * 16;
  const float* qrow = q + (size_t)bh * N_ * D_ + (size_t)n0 * D_;

  // stage q tile (bf16) — wave-private, LDS in-order per wave, no block barrier
  #pragma unroll
  for (int i = 0; i < 32; ++i) {
    int li = lane + 32 * i;
    Q[w][li] = (bf16)qrow[li];
  }
  // diag per row (f32 from global)
  {
    const float* p = qrow + lo * 64 + hi * 32;
    float s2 = 0.f;
    #pragma unroll
    for (int j = 0; j < 32; ++j) { float x = p[j]; s2 += x * x; }
    s2 += __shfl_xor(s2, 16);
    DG[w][lo] = s2 * (0.5f * NORM2_);   // lanes l and l^16 write identical value
  }

  const bf16* arow = &Q[w][lo * 64];
  v16bf a0 = a_frag(arow, hi);
  v16bf a1 = a_frag(arow + 32, hi);

  v8f acc[16];
  #pragma unroll
  for (int ct = 0; ct < 16; ++ct) {
    const bf16* bp = projb + (ct * 16 + lo) * 64 + hi * 16;
    v8f c = (v8f){0,0,0,0,0,0,0,0};
    c = wmma_bf16(a0, b_frag(bp), c);
    c = wmma_bf16(a1, b_frag(bp + 32), c);
    acc[ct] = c;
  }
  // scale by normalizer, per-row max over m, exp(dash - diag - rowmax)
  float rm[8];
  #pragma unroll
  for (int r = 0; r < 8; ++r) rm[r] = -3.4e38f;
  #pragma unroll
  for (int ct = 0; ct < 16; ++ct)
    #pragma unroll
    for (int r = 0; r < 8; ++r) {
      acc[ct][r] *= NORMALIZER_;
      rm[r] = fmaxf(rm[r], acc[ct][r]);
    }
  #pragma unroll
  for (int r = 0; r < 8; ++r) {      // reduce across the 16 lanes sharing a row
    rm[r] = fmaxf(rm[r], __shfl_xor(rm[r], 1));
    rm[r] = fmaxf(rm[r], __shfl_xor(rm[r], 2));
    rm[r] = fmaxf(rm[r], __shfl_xor(rm[r], 4));
    rm[r] = fmaxf(rm[r], __shfl_xor(rm[r], 8));
  }
  v4f dg0 = *(const v4f*)&DG[w][hi * 8];
  v4f dg1 = *(const v4f*)&DG[w][hi * 8 + 4];
  #pragma unroll
  for (int ct = 0; ct < 16; ++ct)
    #pragma unroll
    for (int r = 0; r < 8; ++r) {
      float dgv = (r < 4) ? dg0[r] : dg1[r - 4];
      acc[ct][r] = __expf(acc[ct][r] - dgv - rm[r]);
    }

  // GEMM2: O[16x64] += qp[16x256] @ ctxT^T, K chunks of 32 (two acc tiles)
  v8f O[4];
  #pragma unroll
  for (int et = 0; et < 4; ++et) O[et] = (v8f){0,0,0,0,0,0,0,0};
  const bf16* ctxb = ctxT + (size_t)bh * D_ * M_;
  #pragma unroll
  for (int kt = 0; kt < 8; ++kt) {
    #pragma unroll
    for (int half = 0; half < 2; ++half) {
      int ct = kt * 2 + half;
      #pragma unroll
      for (int r = 0; r < 8; ++r)
        QP[w][(r + 8 * hi) * 40 + half * 16 + lo] = (bf16)acc[ct][r];
    }
    v16bf a = a_frag(&QP[w][lo * 40], hi);
    #pragma unroll
    for (int et = 0; et < 4; ++et) {
      const bf16* bp = ctxb + (size_t)(et * 16 + lo) * M_ + kt * 32 + hi * 16;
      O[et] = wmma_bf16(a, b_frag(bp), O[et]);
    }
  }
  // out = ratio * (O + EPS * ctxsum[e])
  float* ob = out + (size_t)bh * N_ * D_ + (size_t)n0 * D_;
  #pragma unroll
  for (int et = 0; et < 4; ++et) {
    float cs = ctxsum[bh * 64 + et * 16 + lo];
    #pragma unroll
    for (int r = 0; r < 8; ++r)
      ob[(r + 8 * hi) * 64 + et * 16 + lo] = RATIO_ * (O[et][r] + EPSV_ * cs);
  }
}

// ---------------------------------------------------------------------------
extern "C" void kernel_launch(void* const* d_in, const int* in_sizes, int n_in,
                              void* d_out, int out_size, void* d_ws, size_t ws_size,
                              hipStream_t stream) {
  const float* qp  = (const float*)d_in[0];
  const float* kp  = (const float*)d_in[1];
  const float* vp  = (const float*)d_in[2];
  const float* pj  = (const float*)d_in[3];
  float* outp = (float*)d_out;

  char* ws = (char*)d_ws;
  size_t off = 0;
  bf16*  projb    = (bf16*)(ws + off);  off += (size_t)M_ * D_ * 2;            // 32 KB
  bf16*  ctxT     = (bf16*)(ws + off);  off += (size_t)BH_ * D_ * M_ * 2;      // 2 MB
  float* ctxsum   = (float*)(ws + off); off += (size_t)BH_ * 64 * 4;           // 16 KB
  float* chunkmax = (float*)(ws + off); off += (size_t)BH_ * NCH_ * 4;         // 1 KB
  float* vsumpart = (float*)(ws + off); off += (size_t)BH_ * NCH_ * 64 * 4;    // 64 KB
  float* ctxpart  = (float*)(ws + off); off += (size_t)BH_ * NCH_ * M_ * D_ * 4; // 16 MB
  (void)in_sizes; (void)n_in; (void)out_size; (void)ws_size;

  k_prep<<<dim3(64), 256, 0, stream>>>(pj, projb);
  k_ctx <<<dim3(BH_, NCH_), 256, 0, stream>>>(kp, vp, projb, ctxpart, vsumpart, chunkmax);
  k_red <<<dim3(BH_), 256, 0, stream>>>(ctxpart, vsumpart, chunkmax, ctxT, ctxsum);
  k_out <<<dim3(BH_, N_ / 128), 256, 0, stream>>>(qp, projb, ctxT, ctxsum, outp);
}
